// FullRecurrenceCell_29497835389001
// MI455X (gfx1250) — matrix operand order repaired and checked by reading, not compiled, not run on hardware
//
#include <hip/hip_runtime.h>

// Problem dims (fixed by reference)
#define T_DIM 1024
#define B_DIM 16
#define D_DIM 1024
#define TBD   (T_DIM * B_DIM * D_DIM)          // 16,777,216
#define HTOT  ((T_DIM + 1) * B_DIM * D_DIM)    // 16,793,600

typedef __attribute__((ext_vector_type(16))) __bf16   v16bf;
typedef __attribute__((ext_vector_type(8)))  float    v8f;
typedef __attribute__((ext_vector_type(8)))  unsigned v8u;
typedef __attribute__((ext_vector_type(4)))  int      v4i;

typedef __attribute__((address_space(1))) v4i* gptr_v4i;  // global int4*
typedef __attribute__((address_space(3))) v4i* lptr_v4i;  // LDS int4*
typedef __attribute__((address_space(3))) char lds_char;

// ---------------------------------------------------------------------------
// CDNA5 async global->LDS copy (16B per lane), ASYNCcnt-tracked.
// ---------------------------------------------------------------------------
__device__ __forceinline__ void async_g2l_b128(const void* g, void* l) {
#if __has_builtin(__builtin_amdgcn_global_load_async_to_lds_b128)
    __builtin_amdgcn_global_load_async_to_lds_b128(
        (gptr_v4i)(void*)(size_t)g, (lptr_v4i)l, 0, 0);
#else
    unsigned laddr = (unsigned)(size_t)(lds_char*)l;
    asm volatile("global_load_async_to_lds_b128 %0, %1, off"
                 :: "v"(laddr), "v"(g) : "memory");
#endif
}

__device__ __forceinline__ void wait_async0() {
#if __has_builtin(__builtin_amdgcn_s_wait_asynccnt)
    __builtin_amdgcn_s_wait_asynccnt(0);
#else
    asm volatile("s_wait_asynccnt 0x0" ::: "memory");
#endif
}

// ---------------------------------------------------------------------------
// Init: zero barrier counter, copy h0 into h_out[0] (f32) and bf16 slot 0.
// ---------------------------------------------------------------------------
__global__ __launch_bounds__(256) void rnn_init_kernel(const float* __restrict__ h0,
                                                       float* __restrict__ h_out,
                                                       __bf16* __restrict__ hbf_g,
                                                       unsigned* __restrict__ ctr) {
    int i = blockIdx.x * 256 + threadIdx.x;
    if (i == 0) *ctr = 0u;
    if (i < B_DIM * D_DIM) {
        float v = h0[i];
        h_out[i] = v;
        hbf_g[i] = (__bf16)v;
    }
}

// ---------------------------------------------------------------------------
// Projection GEMM: out[m, n] = act( sum_d x[m, d] * W[n, d] + bias[n] )
// Block = 256 threads (8 waves), tile = 64 rows x 128 cols; bf16 WMMA, f32 acc.
// ---------------------------------------------------------------------------
__global__ __launch_bounds__(256) void proj_gemm_kernel(const float* __restrict__ x,
                                                        const float* __restrict__ W,
                                                        const float* __restrict__ bias,
                                                        float* __restrict__ out,
                                                        int apply_sigmoid) {
    const int wave = threadIdx.x >> 5;
    const int lane = threadIdx.x & 31;
    const int l    = lane & 15;
    const int hi   = lane >> 4;

    const int m0 = blockIdx.x * 64 + (wave & 3) * 16;   // 256 m-blocks of 64
    const int n0 = blockIdx.y * 128 + (wave >> 2) * 64; // 8 n-blocks of 128

    v8f acc[4];
#pragma unroll
    for (int nt = 0; nt < 4; ++nt)
#pragma unroll
        for (int r = 0; r < 8; ++r) acc[nt][r] = 0.0f;

    const float* xrow = x + (size_t)(m0 + l) * D_DIM;

    for (int k0 = 0; k0 < D_DIM; k0 += 32) {
        // A fragment: 16x32 bf16, row M = l, K pattern per ISA table.
        v16bf a;
#pragma unroll
        for (int v = 0; v < 8; ++v) {
            int ka = k0 + ((v >> 2) << 4) + hi * 8 + ((v & 3) << 1);
            a[2 * v]     = (__bf16)xrow[ka];
            a[2 * v + 1] = (__bf16)xrow[ka + 1];
        }
#pragma unroll
        for (int nt = 0; nt < 4; ++nt) {
            // B fragment: 32x16 bf16, B[k][n] = W[n][k]; lane = column n.
            const float* wrow = W + (size_t)(n0 + nt * 16 + l) * D_DIM;
            v16bf bm;
#pragma unroll
            for (int v = 0; v < 8; ++v) {
                int kb = k0 + ((v >> 2) << 4) + hi * 8 + ((v & 3) << 1);
                bm[2 * v]     = (__bf16)wrow[kb];
                bm[2 * v + 1] = (__bf16)wrow[kb + 1];
            }
            acc[nt] = __builtin_amdgcn_wmma_f32_16x16x32_bf16(
                false, a, false, bm, (short)0, acc[nt], false, false);
        }
    }

    // Epilogue: bias + optional sigmoid. C/D layout: VGPR r -> row r + hi*8.
#pragma unroll
    for (int nt = 0; nt < 4; ++nt) {
        int n = n0 + nt * 16 + l;
        float bs = bias[n];
#pragma unroll
        for (int r = 0; r < 8; ++r) {
            int m = m0 + r + hi * 8;
            float v = acc[nt][r] + bs;
            if (apply_sigmoid) v = 1.0f / (1.0f + __expf(-v));
            out[(size_t)m * D_DIM + n] = v;
        }
    }
}

// ---------------------------------------------------------------------------
// Persistent recurrence: 16 blocks x 128 threads (4 waves).
// Block j owns output columns [j*64, j*64+64). The R_h slice is converted to
// bf16 WMMA B-fragments held in REGISTERS (32 x v16bf = 256 VGPRs/wave; CDNA5
// wave32 allows 1024, and we run 1 wave/SIMD). h state: f32 in L2 (output)
// plus a bf16 shadow (ws, double-buffered) async-copied into LDS each step.
// Cross-block sync: agent-scope monotonic atomic barrier.
// ---------------------------------------------------------------------------
#define LROW 1032  // 1024 + 8 bf16 pad; row pitch 2064 B (16B-aligned)
#define SMEM_BYTES ((64 * LROW + B_DIM * LROW) * 2)

__global__ __launch_bounds__(128) void rnn_recurrence_kernel(
    const float* __restrict__ x,         // [T,B,D]
    const float* __restrict__ cand_in,   // [T,B,D]  x@W_x^T + b
    const float* __restrict__ delta_all, // [T,B,D]  sigmoid(x@W_delta^T + b_delta)
    const float* __restrict__ R_h,       // [D,D]
    const float* __restrict__ b_gate,    // [D]
    float* __restrict__ out,             // [T,B,D]
    float* __restrict__ h_out,           // [T+1,B,D] (f32 outputs)
    __bf16* __restrict__ hbf_g,          // [2][B,D] bf16 shadow of h
    unsigned* __restrict__ ctr) {
    extern __shared__ char smem[];
    __bf16* Rh  = (__bf16*)smem;                      // [64][LROW] (init only)
    __bf16* hbf = (__bf16*)smem + 64 * LROW;          // [B_DIM][LROW]

    const int j    = blockIdx.x;        // column-slice owner
    const int tid  = threadIdx.x;
    const int wave = tid >> 5;
    const int lane = tid & 31;
    const int l    = lane & 15;
    const int hi   = lane >> 4;

    // One-time: load this block's R_h slice (rows j*64..j*64+63) as bf16.
    for (int c = tid; c < (64 * D_DIM) / 4; c += 128) {
        int i4 = c << 2;
        int e = i4 >> 10, d = i4 & (D_DIM - 1);
        float4 f = ((const float4*)R_h)[((size_t)(j * 64 + e) * D_DIM + d) >> 2];
        __bf16* dst = Rh + e * LROW + d;
        dst[0] = (__bf16)f.x; dst[1] = (__bf16)f.y;
        dst[2] = (__bf16)f.z; dst[3] = (__bf16)f.w;
    }
    __syncthreads();

    const int e_loc  = wave * 16 + l;                 // local column for B frag
    const int e_glob = j * 64 + wave * 16 + l;        // global output column
    const float bg   = b_gate[e_glob];

    // One-time: hoist all 32 time-invariant B fragments into registers.
    v16bf bfrag[32];
#pragma unroll
    for (int kk = 0; kk < 32; ++kk) {
        v8u bu;
#pragma unroll
        for (int v = 0; v < 8; ++v) {
            int kb = (kk << 5) + ((v >> 2) << 4) + hi * 8 + ((v & 3) << 1);
            bu[v] = *(const unsigned*)(Rh + e_loc * LROW + kb);
        }
        bfrag[kk] = __builtin_bit_cast(v16bf, bu);
    }

#pragma unroll 1
    for (unsigned t = 0; t < T_DIM; ++t) {
        // Async-stage bf16 h[t] (32 KB) from its global shadow into LDS.
        const __bf16* hsrc_bf = hbf_g + (size_t)(t & 1) * (B_DIM * D_DIM);
#pragma unroll
        for (int k = 0; k < 16; ++k) {
            int c  = tid + k * 128;            // 16B chunk id (8 bf16)
            int bb = c >> 7;                   // batch row
            int d  = (c & 127) << 3;           // column within row
            async_g2l_b128(hsrc_bf + (size_t)bb * D_DIM + d,
                           hbf + bb * LROW + d);
        }
        wait_async0();
        __syncthreads();

        // acc = (h[t] @ R_h^T)[:, tile], K = 1024 in 32 bf16-WMMA steps.
        // Steady state: only A fragments come from LDS; B lives in VGPRs.
        v8f acc;
#pragma unroll
        for (int r = 0; r < 8; ++r) acc[r] = 0.0f;

#pragma unroll
        for (int kk = 0; kk < 32; ++kk) {
            v8u au;
#pragma unroll
            for (int v = 0; v < 8; ++v) {
                int ka = (kk << 5) + ((v >> 2) << 4) + hi * 8 + ((v & 3) << 1);
                au[v] = *(const unsigned*)(hbf + l * LROW + ka);
            }
            v16bf a = __builtin_bit_cast(v16bf, au);
            acc = __builtin_amdgcn_wmma_f32_16x16x32_bf16(
                false, a, false, bfrag[kk], (short)0, acc, false, false);
        }

        // Fused elementwise: candidate, delta blend, silu gate, outputs.
        const float* hsrc = h_out + (size_t)t * (B_DIM * D_DIM);
        __bf16* hdst_bf = hbf_g + (size_t)((t + 1) & 1) * (B_DIM * D_DIM);
#pragma unroll
        for (int r = 0; r < 8; ++r) {
            int bb = r + hi * 8;  // batch row
            size_t idx = ((size_t)t * B_DIM + bb) * D_DIM + e_glob;
            float hprev = hsrc[(size_t)bb * D_DIM + e_glob];
            float cand  = tanhf(cand_in[idx] + acc[r]);
            float dlt   = delta_all[idx];
            float hn    = (1.0f - dlt) * hprev + dlt * cand;
            float g     = hn + x[idx] + bg;
            float gate  = g / (1.0f + __expf(-g));  // silu
            out[idx] = hn * gate;
            h_out[(size_t)(t + 1) * (B_DIM * D_DIM) + (size_t)bb * D_DIM + e_glob] = hn;
            hdst_bf[(size_t)bb * D_DIM + e_glob] = (__bf16)hn;
        }

        // Device-scope barrier across the 16 blocks (monotonic counter).
        __threadfence();
        __syncthreads();
        if (tid == 0) {
            __hip_atomic_fetch_add(ctr, 1u, __ATOMIC_RELEASE, __HIP_MEMORY_SCOPE_AGENT);
            while (__hip_atomic_load(ctr, __ATOMIC_ACQUIRE, __HIP_MEMORY_SCOPE_AGENT) <
                   16u * (t + 1))
                __builtin_amdgcn_s_sleep(2);
        }
        __syncthreads();
    }
}

// ---------------------------------------------------------------------------
// Host entry
// ---------------------------------------------------------------------------
extern "C" void kernel_launch(void* const* d_in, const int* in_sizes, int n_in,
                              void* d_out, int out_size, void* d_ws, size_t ws_size,
                              hipStream_t stream) {
    (void)in_sizes; (void)n_in; (void)out_size; (void)ws_size;

    const float* x       = (const float*)d_in[0];
    const float* h0      = (const float*)d_in[1];
    const float* W_x     = (const float*)d_in[2];
    const float* R_h     = (const float*)d_in[3];
    const float* W_delta = (const float*)d_in[4];
    const float* b       = (const float*)d_in[5];
    const float* b_delta = (const float*)d_in[6];
    const float* b_gate  = (const float*)d_in[7];

    float* out   = (float*)d_out;          // [T,B,D]
    float* h_out = out + (size_t)TBD;      // [T+1,B,D]

    // Workspace layout: ctr | bf16 h shadow (2 slots) | delta_all | cand_in
    unsigned* ctr    = (unsigned*)d_ws;
    __bf16* hbf_g    = (__bf16*)((char*)d_ws + 4096);
    float* delta_all = (float*)((char*)d_ws + 4096 + 2 * B_DIM * D_DIM * sizeof(__bf16));
    float* cand_in   = delta_all + (size_t)TBD;

    (void)hipFuncSetAttribute((const void*)rnn_recurrence_kernel,
                              hipFuncAttributeMaxDynamicSharedMemorySize, SMEM_BYTES);

    rnn_init_kernel<<<64, 256, 0, stream>>>(h0, h_out, hbf_g, ctr);

    dim3 pgrid(256, 8);
    proj_gemm_kernel<<<pgrid, 256, 0, stream>>>(x, W_x,     b,       cand_in,   0);
    proj_gemm_kernel<<<pgrid, 256, 0, stream>>>(x, W_delta, b_delta, delta_all, 1);

    rnn_recurrence_kernel<<<16, 128, SMEM_BYTES, stream>>>(
        x, cand_in, delta_all, R_h, b_gate, out, h_out, hbf_g, ctr);
}
